// HierarchicalMoE_43688407335206
// MI455X (gfx1250) — matrix operand order repaired and checked by reading, not compiled, not run on hardware
//
#include <hip/hip_runtime.h>
#include <hip/hip_bf16.h>
#include <math.h>

// ---------------------------------------------------------------------------
// Hierarchical MoE, MI455X (gfx1250, wave32, WMMA).
// Sparse (routed) compute: ~69 GFLOP bf16 vs 550 GFLOP dense fp32.
// HBM floor ~300MB @23.3TB/s ~= 13us; bf16 WMMA compute ~= 14us -> balanced.
// This revision: native __bf16 casts (v_cvt_pk_bf16_f32) instead of manual
// integer RNE, cutting the VALU work between weight loads and v_wmma.
// ---------------------------------------------------------------------------

#define D_MODEL 1024
#define D_FF    2048
#define NEXP    16
#define NTOK    4096
#define CAP     1024   // per-expert token capacity materialized in ws

typedef __attribute__((ext_vector_type(16))) __bf16        v16bf;
typedef __attribute__((ext_vector_type(8)))  float         v8f;
typedef __attribute__((ext_vector_type(8)))  unsigned short ush8;

union ABf { v16bf v; __bf16 b[16]; ush8 h[2]; };

__device__ __forceinline__ float gelu_exact(float v) {
  return 0.5f * v * (1.0f + erff(v * 0.70710678118654752440f));
}

// ---------------------------------------------------------------------------
// Kernel 0: zero output (atomic-accumulated) and per-expert counters.
// ---------------------------------------------------------------------------
__global__ void moe_init(float* __restrict__ out, int* __restrict__ counts) {
  size_t i = (size_t)blockIdx.x * blockDim.x + threadIdx.x;
  if (i < (size_t)NTOK * D_MODEL) out[i] = 0.0f;
  if (blockIdx.x == 0 && threadIdx.x < NEXP) counts[threadIdx.x] = 0;
}

// ---------------------------------------------------------------------------
// Kernel 1: router. One wave32 per token (8 tokens / 256-thread block).
// Implements the reference's two-level gate, thresholds, top-2 fallback,
// and renormalization exactly (fp32).
// ---------------------------------------------------------------------------
__global__ void __launch_bounds__(256) moe_router(
    const float* __restrict__ x, const float* __restrict__ Wg,
    const float* __restrict__ We, float* __restrict__ wts,
    int* __restrict__ counts, int* __restrict__ list) {
  const int lane = threadIdx.x & 31;
  const int t = blockIdx.x * 8 + (threadIdx.x >> 5);

  float acc[20];
#pragma unroll
  for (int i = 0; i < 20; ++i) acc[i] = 0.0f;

  for (int d = lane; d < D_MODEL; d += 32) {
    float xv = x[(size_t)t * D_MODEL + d];
#pragma unroll
    for (int g = 0; g < 4; ++g)  acc[g]     += xv * Wg[d * 4  + g];
#pragma unroll
    for (int e = 0; e < 16; ++e) acc[4 + e] += xv * We[d * 16 + e];
  }
#pragma unroll
  for (int i = 0; i < 20; ++i) {
#pragma unroll
    for (int off = 16; off > 0; off >>= 1) acc[i] += __shfl_xor(acc[i], off, 32);
  }
  if (lane != 0) return;

  // Level 1: group softmax + threshold
  float gp[4];
  float gmx = -1e30f;
  for (int g = 0; g < 4; ++g) gmx = fmaxf(gmx, acc[g]);
  float gs = 0.0f;
  for (int g = 0; g < 4; ++g) { gp[g] = expf(acc[g] - gmx); gs += gp[g]; }
  for (int g = 0; g < 4; ++g) gp[g] /= gs;
  bool gmask[4];
  for (int g = 0; g < 4; ++g) gmask[g] = (gp[g] >= 0.25f);

  // Level 2: per-group expert softmax + threshold
  float fp[16]; bool vmask[16];
  for (int g = 0; g < 4; ++g) {
    float m = -1e30f;
    for (int j = 0; j < 4; ++j) m = fmaxf(m, acc[4 + g * 4 + j]);
    float ep[4], s = 0.0f;
    for (int j = 0; j < 4; ++j) { ep[j] = expf(acc[4 + g * 4 + j] - m); s += ep[j]; }
    for (int j = 0; j < 4; ++j) {
      float p = ep[j] / s;
      fp[g * 4 + j] = gp[g] * p;
      vmask[g * 4 + j] = gmask[g] && (p >= 0.25f);
    }
  }
  int nsel = 0;
  for (int e = 0; e < 16; ++e) nsel += vmask[e] ? 1 : 0;

  // top-2 by prob, ties -> lower index (matches jax.lax.top_k)
  int i1 = 0; float bv = -1e30f;
  for (int e = 0; e < 16; ++e) if (fp[e] > bv) { bv = fp[e]; i1 = e; }
  int i2 = (i1 == 0) ? 1 : 0; bv = -1e30f;
  for (int e = 0; e < 16; ++e) if (e != i1 && fp[e] > bv) { bv = fp[e]; i2 = e; }

  const bool use_topk = (nsel < 2);
  float sel[16], ssum = 0.0f;
  for (int e = 0; e < 16; ++e) {
    bool m = use_topk ? (e == i1 || e == i2) : vmask[e];
    sel[e] = m ? fp[e] : 0.0f;
    ssum += sel[e];
  }
  ssum = fmaxf(ssum, 1e-9f);
  for (int e = 0; e < 16; ++e) {
    float w = sel[e] / ssum;
    wts[t * 16 + e] = w;
    if (w > 0.0f) {
      int slot = atomicAdd(&counts[e], 1);
      if (slot < CAP) list[e * CAP + slot] = t;
    }
  }
}

// ---------------------------------------------------------------------------
// Kernel 2: H = gelu(Xe @ w1[e] + b1[e]), bf16 WMMA, H stored bf16 in ws.
// Block = 256 thr (8 waves), tile: M=64 tokens x N=128 ff-cols per fb step.
// Wave wv owns 16 ff columns; 4 WMMA M-tiles per wave. K = D_MODEL = 1024.
// ---------------------------------------------------------------------------
__global__ void __launch_bounds__(256) moe_ffn1(
    const float* __restrict__ x, const float* __restrict__ w1,
    const float* __restrict__ b1, const int* __restrict__ counts,
    const int* __restrict__ list, __bf16* __restrict__ H) {
  const int e = blockIdx.x, mb = blockIdx.y;
  int cnt = counts[e]; if (cnt > CAP) cnt = CAP;
  if (mb * 64 >= cnt) return;                       // uniform early-out

  const int tid = threadIdx.x, wv = tid >> 5, lane = tid & 31;
  const int lh = lane >> 4, ln = lane & 15;

  // X K-slice staged as bf16: [kchunk(4)][row(64)][8 halves] -> ds_load_b128
  __shared__ __attribute__((aligned(16))) __bf16 Xs[4 * 64 * 8];
  __shared__ int Toks[64];
  if (tid < 64) {
    int g = mb * 64 + tid;
    Toks[tid] = (g < cnt) ? list[e * CAP + g] : 0;  // pad rows read token 0
  }
  __syncthreads();

  const float* w1e = w1 + (size_t)e * D_MODEL * D_FF;

  for (int fb = 0; fb < D_FF / 128; ++fb) {
    const int f = fb * 128 + wv * 16 + ln;
    v8f acc[4] = {};
    for (int ks = 0; ks < D_MODEL / 32; ++ks) {
      __syncthreads();
      { // cooperative gather: 64 rows x 32 dims -> bf16 LDS (packed cvt)
        int row = tid & 63, c = tid >> 6;
        const float* xp = x + (size_t)Toks[row] * D_MODEL + ks * 32 + c * 8;
        __bf16* dst = &Xs[(c * 64 + row) * 8];
#pragma unroll
        for (int i = 0; i < 8; ++i) dst[i] = (__bf16)xp[i];
      }
      __syncthreads();

      // B fragment: element j -> K = (lane>>4)*16 + j, N = lane&15
      ABf B;
      const int kk = ks * 32 + lh * 16;
#pragma unroll
      for (int j = 0; j < 16; ++j)
        B.b[j] = (__bf16)w1e[(size_t)(kk + j) * D_FF + f];

#pragma unroll
      for (int mt = 0; mt < 4; ++mt) {
        // A fragment: M = lane&15; K = (lane>>4)*16 + j
        int row = mt * 16 + ln;
        ABf A;
        A.h[0] = *(const ush8*)&Xs[((2 * lh)     * 64 + row) * 8];
        A.h[1] = *(const ush8*)&Xs[((2 * lh + 1) * 64 + row) * 8];
        acc[mt] = __builtin_amdgcn_wmma_f32_16x16x32_bf16(
            false, A.v, false, B.v, (short)0, acc[mt], false, false);
      }
    }
    // epilogue: bias + exact gelu, store bf16 H
    const float bias = b1[e * D_FF + f];
#pragma unroll
    for (int mt = 0; mt < 4; ++mt) {
#pragma unroll
      for (int r = 0; r < 8; ++r) {          // C: M = r + (lane>>4)*8
        int grow = mb * 64 + mt * 16 + r + lh * 8;
        float hv = gelu_exact(acc[mt][r] + bias);
        H[(size_t)(e * CAP + grow) * D_FF + f] = (__bf16)hv;
      }
    }
  }
}

// ---------------------------------------------------------------------------
// Kernel 3: Y = H @ w2[e] + b2[e]; out[tok] += weight[tok,e] * Y.
// Same tiling; K = D_FF = 2048. A is bf16-contiguous -> direct 16B loads.
// ---------------------------------------------------------------------------
__global__ void __launch_bounds__(256) moe_ffn2(
    const __bf16* __restrict__ H, const float* __restrict__ w2,
    const float* __restrict__ b2, const float* __restrict__ wts,
    const int* __restrict__ counts, const int* __restrict__ list,
    float* __restrict__ out) {
  const int e = blockIdx.x, mb = blockIdx.y;
  int cnt = counts[e]; if (cnt > CAP) cnt = CAP;
  if (mb * 64 >= cnt) return;

  const int tid = threadIdx.x, wv = tid >> 5, lane = tid & 31;
  const int lh = lane >> 4, ln = lane & 15;

  __shared__ int   Toks[64];
  __shared__ float Wt[64];
  if (tid < 64) {
    int g = mb * 64 + tid;
    int tk = (g < cnt) ? list[e * CAP + g] : 0;
    Toks[tid] = tk;
    Wt[tid]   = (g < cnt) ? wts[tk * 16 + e] : 0.0f;  // pad rows: weight 0
  }
  __syncthreads();

  const float* w2e = w2 + (size_t)e * D_FF * D_MODEL;
  const __bf16* He = H + (size_t)(e * CAP + mb * 64) * D_FF;

  for (int db = 0; db < D_MODEL / 128; ++db) {
    const int d = db * 128 + wv * 16 + ln;
    v8f acc[4] = {};
    for (int ks = 0; ks < D_FF / 32; ++ks) {
      ABf B;
      const int kk = ks * 32 + lh * 16;
#pragma unroll
      for (int j = 0; j < 16; ++j)
        B.b[j] = (__bf16)w2e[(size_t)(kk + j) * D_MODEL + d];
#pragma unroll
      for (int mt = 0; mt < 4; ++mt) {
        const __bf16* hp = He + (size_t)(mt * 16 + ln) * D_FF + ks * 32;
        ABf A;
        A.h[0] = *(const ush8*)(hp + lh * 16);
        A.h[1] = *(const ush8*)(hp + lh * 16 + 8);
        acc[mt] = __builtin_amdgcn_wmma_f32_16x16x32_bf16(
            false, A.v, false, B.v, (short)0, acc[mt], false, false);
      }
    }
    const float bias = b2[e * D_MODEL + d];
#pragma unroll
    for (int mt = 0; mt < 4; ++mt) {
#pragma unroll
      for (int r = 0; r < 8; ++r) {
        int row = mt * 16 + r + lh * 8;
        float wt = Wt[row];
        if (wt != 0.0f)
          atomicAdd(&out[(size_t)Toks[row] * D_MODEL + d],
                    (acc[mt][r] + bias) * wt);
      }
    }
  }
}

// ---------------------------------------------------------------------------
// Launch. ws layout (<= ~67.5 MB):
//   [0, 256KB)      router weights  float[4096][16]
//   [+256B]         counts          int[16]
//   [+64KB]         token lists     int[16][CAP]
//   [+64MB]         H (bf16)        __bf16[16*CAP][D_FF]
// ---------------------------------------------------------------------------
extern "C" void kernel_launch(void* const* d_in, const int* in_sizes, int n_in,
                              void* d_out, int out_size, void* d_ws, size_t ws_size,
                              hipStream_t stream) {
  (void)in_sizes; (void)n_in; (void)out_size; (void)ws_size;
  const float* x  = (const float*)d_in[0];
  const float* Wg = (const float*)d_in[1];
  const float* We = (const float*)d_in[2];
  const float* w1 = (const float*)d_in[3];
  const float* b1 = (const float*)d_in[4];
  const float* w2 = (const float*)d_in[5];
  const float* b2 = (const float*)d_in[6];
  float* out = (float*)d_out;

  char* ws = (char*)d_ws;
  float* wts    = (float*)ws;                                   // 256 KB
  int*   counts = (int*)(ws + 262144);                          // 64 B
  int*   list   = (int*)(ws + 262144 + 256);                    // 64 KB
  __bf16* H     = (__bf16*)(ws + 262144 + 256 + 65536);         // 64 MB

  moe_init  <<<(NTOK * D_MODEL + 255) / 256, 256, 0, stream>>>(out, counts);
  moe_router<<<NTOK / 8, 256, 0, stream>>>(x, Wg, We, wts, counts, list);
  moe_ffn1  <<<dim3(NEXP, CAP / 64), 256, 0, stream>>>(x, w1, b1, counts, list, H);
  moe_ffn2  <<<dim3(NEXP, CAP / 64), 256, 0, stream>>>(H, w2, b2, wts, counts, list, out);
}
